// DRNN_8435315769657
// MI455X (gfx1250) — compile-verified, matrix operand order checked by reading, not run on hardware
//
#include <hip/hip_runtime.h>
#include <hip/hip_bf16.h>
#include <math.h>

// ---------------------------------------------------------------------------
// Types for CDNA5 WMMA (wave32). bf16 16x16x32 -> f32 accumulate.
// ---------------------------------------------------------------------------
typedef __bf16 bf16;
typedef __attribute__((ext_vector_type(16))) __bf16        v16bf;
typedef __attribute__((ext_vector_type(8)))  float         v8f;
typedef __attribute__((ext_vector_type(4)))  unsigned int  u32x4;

union Frag16 { u32x4 q[2]; v16bf v; };

__device__ __forceinline__ v8f wmma_bf16(v16bf a, v16bf b, v8f c) {
  return __builtin_amdgcn_wmma_f32_16x16x32_bf16(
      /*neg_a=*/false, a, /*neg_b=*/false, b,
      /*c_mod=*/(short)0, c, /*reuse_a=*/false, /*reuse_b=*/false);
}

// Branch-free activations for the recurrence critical path:
// v_exp_f32 + v_rcp_f32 only, no libm slow-path exec-mask divergence.
__device__ __forceinline__ float fast_rcp(float x) { return __builtin_amdgcn_rcpf(x); }
__device__ __forceinline__ float fast_sigmoid(float x) {
  return fast_rcp(1.0f + __expf(-x));
}
__device__ __forceinline__ float fast_tanh(float x) {
  float e = __expf(-2.0f * fabsf(x));          // e in (0,1]
  float r = (1.0f - e) * fast_rcp(1.0f + e);   // tanh(|x|)
  return copysignf(r, x);
}
__device__ __forceinline__ float geluf_(float x) {
  return 0.5f * x * (1.0f + erff(x * 0.70710678118654752440f));
}

// LSTM activation panel: 32 rows x [x_t(0:64) | h0(64:320) | h1(320:576) | h2(576:832)]
// stride 840 bf16 (=1680B) -> 16 rows map to disjoint 4-bank groups.
#define ASTR 840

// ---------------------------------------------------------------------------
// Weight packing: W (dout x din, row-major f32) -> bf16 WMMA B-fragments.
// Fragment layout: dst[((nj*KT + kt)*32 + lane)*16 + e], where for lane:
//   col   = nj*16 + (lane&15)
//   k     = kt*32 + (lane>>4)*8 + (e>>3)*16 + (e&7)     (matches 16-bit B layout)
// LSTM version packs [Wih | Whh] concatenated along K.
// ---------------------------------------------------------------------------
__global__ void pack_lstm_kernel(const float* __restrict__ wih,
                                 const float* __restrict__ whh,
                                 int din, int total, bf16* __restrict__ dst) {
  int idx = blockIdx.x * 256 + threadIdx.x;
  if (idx >= total) return;
  int K = din + 256, KT = K >> 5;
  int e = idx & 15;
  int t1 = idx >> 4;
  int lane = t1 & 31;
  int t2 = t1 >> 5;
  int kt = t2 % KT;
  int nj = t2 / KT;
  int col = nj * 16 + (lane & 15);
  int k = kt * 32 + ((lane >> 4) * 8) + ((e >> 3) * 16) + (e & 7);
  float v = (k < din) ? wih[col * din + k] : whh[col * 256 + (k - din)];
  dst[idx] = (bf16)v;
}

__global__ void pack_w_kernel(const float* __restrict__ w, int K, int total,
                              bf16* __restrict__ dst) {
  int idx = blockIdx.x * 256 + threadIdx.x;
  if (idx >= total) return;
  int KT = K >> 5;
  int e = idx & 15;
  int t1 = idx >> 4;
  int lane = t1 & 31;
  int t2 = t1 >> 5;
  int kt = t2 % KT;
  int nj = t2 / KT;
  int col = nj * 16 + (lane & 15);
  int k = kt * 32 + ((lane >> 4) * 8) + ((e >> 3) * 16) + (e & 7);
  dst[idx] = (bf16)w[col * K + k];
}

__global__ void add_bias_kernel(const float* __restrict__ a,
                                const float* __restrict__ b,
                                float* __restrict__ d) {
  int i = blockIdx.x * 256 + threadIdx.x;
  if (i < 1024) d[i] = a[i] + b[i];
}

// ---------------------------------------------------------------------------
// One fused LSTM gate-GEMM + cell update step.
// 32 waves: wave w -> mi=w>>4 (batch tile), r=w&15 (unit group). Its 4 N-tiles
// are the 4 gates of units [r*16, r*16+16): nj = gate*16 + r, so gate column
// = gate*256 + r*16 + (lane&15). i/f/g/o for a unit sit in the SAME lane/VGPR
// slot across the 4 accumulators -> cell update is pure per-lane VALU, cell
// state c stays in 8 VGPRs/wave (fp32). No gate LDS needed.
//
// NOTE: the kt loop is kept ROLLED (#pragma unroll 1). Full unroll made the
// compiler hoist all B-fragment global loads and spill them through scratch;
// rolled, one B fragment is live at a time and latency is hidden by the 32
// resident waves instead of in-thread hoisting.
// ---------------------------------------------------------------------------
__device__ __forceinline__ void lstm_step(
    bf16* __restrict__ aLDS, const bf16* __restrict__ frag,
    const float* __restrict__ bias, int cb, int KT, int hcol,
    int rowA, int khalf8, int uu, int rowv0, int r, int lane,
    v8f& cst, float* __restrict__ snap) {
  __syncthreads();  // inputs (x_t / h writes) visible before WMMA reads
  v8f acc[4];
#pragma unroll
  for (int j = 0; j < 4; ++j) acc[j] = (v8f){0.f, 0.f, 0.f, 0.f, 0.f, 0.f, 0.f, 0.f};
  const int aBase = rowA * ASTR + cb + khalf8;
#pragma unroll 1
  for (int kt = 0; kt < KT; ++kt) {
    Frag16 A;
    const u32x4* ap = reinterpret_cast<const u32x4*>(&aLDS[aBase + kt * 32]);
    A.q[0] = ap[0];   // K = kt*32 + khalf8 + [0..7]
    A.q[1] = ap[2];   // K = kt*32 + khalf8 + 16 + [0..7]
#pragma unroll
    for (int j = 0; j < 4; ++j) {
      const int nj = j * 16 + r;
      Frag16 B;
      const u32x4* bp = reinterpret_cast<const u32x4*>(
          frag + ((size_t)(nj * KT + kt) * 32 + (size_t)lane) * 16);
      B.q[0] = bp[0];
      B.q[1] = bp[1];
      acc[j] = wmma_bf16(A.v, B.v, acc[j]);
    }
  }
  __syncthreads();  // all WMMA reads of h done before h is overwritten below
  const float bi = bias[uu], bf2 = bias[256 + uu], bg = bias[512 + uu], bo = bias[768 + uu];
#pragma unroll
  for (int v = 0; v < 8; ++v) {
    float gi = acc[0][v] + bi;   // C layout: lane l, vgpr v -> row=(l>>4)*8+v, col=l&15
    float gf = acc[1][v] + bf2;
    float gg = acc[2][v] + bg;
    float go = acc[3][v] + bo;
    float c = fast_sigmoid(gf) * cst[v] + fast_sigmoid(gi) * fast_tanh(gg);
    cst[v] = c;
    float h = fast_sigmoid(go) * fast_tanh(c);
    aLDS[(rowv0 + v) * ASTR + hcol + uu] = (bf16)h;
    if (snap) snap[(rowv0 + v) * 2048 + uu] = h;  // F/Bw: [b][8][256]
  }
}

// ---------------------------------------------------------------------------
// Fused dilated 3-layer LSTM, one direction per block (grid=2: fwd, bwd).
// Layer 1 runs at t%8==0, layer 2 at t%64==0; layer-2 state snapshots are the
// only values the rest of the model ever sees (8 distinct tokens / batch).
// ---------------------------------------------------------------------------
__global__ __launch_bounds__(1024) void drnn_kernel(
    const float* __restrict__ x,
    const bf16* __restrict__ ff0, const bf16* __restrict__ ff1, const bf16* __restrict__ ff2,
    const bf16* __restrict__ fb0, const bf16* __restrict__ fb1, const bf16* __restrict__ fb2,
    const float* __restrict__ biasAll, float* __restrict__ Fbase) {
  extern __shared__ char smem[];
  bf16* aLDS = (bf16*)smem;  // 32 x 840 bf16 = 53,760 B
  const int dir = blockIdx.x;
  const bf16* f0 = dir ? fb0 : ff0;
  const bf16* f1 = dir ? fb1 : ff1;
  const bf16* f2 = dir ? fb2 : ff2;
  const float* bias = biasAll + dir * 3072;
  float* snapBase = Fbase + dir * 65536;  // F then Bw, each [32][8][256]

  const int tid = threadIdx.x, lane = tid & 31, w = tid >> 5;
  const int mi = w >> 4, r = w & 15;
  const int rowA = mi * 16 + (lane & 15);
  const int khalf8 = (lane >> 4) * 8;
  const int uu = r * 16 + (lane & 15);
  const int rowv0 = mi * 16 + khalf8;

  for (int i = tid; i < 32 * ASTR; i += 1024) aLDS[i] = (bf16)0.0f;  // h(0)=0
  __syncthreads();

  v8f c0 = {0.f, 0.f, 0.f, 0.f, 0.f, 0.f, 0.f, 0.f};
  v8f c1 = c0, c2 = c0;

#pragma unroll 1
  for (int t = 0; t < 512; ++t) {
    const int tcur = dir ? (511 - t) : t;
#pragma unroll
    for (int k2 = 0; k2 < 2; ++k2) {  // load x_t (nan_to_num) into cols 0..63
      int e = tid + k2 * 1024;
      int bb = e >> 6, col = e & 63;
      float v = x[(bb * 512 + tcur) * 64 + col];
      v = (v == v) ? v : 0.0f;
      aLDS[bb * ASTR + col] = (bf16)v;
    }
    lstm_step(aLDS, f0, bias,        0,   10, 64,  rowA, khalf8, uu, rowv0, r, lane, c0, nullptr);
    if ((t & 7) == 0)
      lstm_step(aLDS, f1, bias + 1024, 64,  16, 320, rowA, khalf8, uu, rowv0, r, lane, c1, nullptr);
    if ((t & 63) == 0)
      lstm_step(aLDS, f2, bias + 2048, 320, 16, 576, rowA, khalf8, uu, rowv0, r, lane, c2,
                snapBase + (t >> 6) * 256);
  }
}

// ---------------------------------------------------------------------------
// Head kernel helpers: WMMA projection of the 8-token panel (rows 8..15 zero).
// kt loop rolled for the same anti-spill reason as lstm_step.
// ---------------------------------------------------------------------------
#define TSTR 520

__device__ __forceinline__ void head_proj(const bf16* __restrict__ tokA,
                                          const bf16* __restrict__ frag,
                                          const float* __restrict__ bias,
                                          float* __restrict__ out, int w, int lane) {
  v8f acc[4];
#pragma unroll
  for (int j = 0; j < 4; ++j) acc[j] = (v8f){0.f, 0.f, 0.f, 0.f, 0.f, 0.f, 0.f, 0.f};
  const int aBase = (lane & 15) * TSTR + ((lane >> 4) * 8);
#pragma unroll 1
  for (int kt = 0; kt < 16; ++kt) {
    Frag16 A;
    const u32x4* ap = reinterpret_cast<const u32x4*>(&tokA[aBase + kt * 32]);
    A.q[0] = ap[0];
    A.q[1] = ap[2];
#pragma unroll
    for (int j = 0; j < 4; ++j) {
      int nj = w * 4 + j;
      Frag16 B;
      const u32x4* bp = reinterpret_cast<const u32x4*>(
          frag + ((size_t)(nj * 16 + kt) * 32 + (size_t)lane) * 16);
      B.q[0] = bp[0];
      B.q[1] = bp[1];
      acc[j] = wmma_bf16(A.v, B.v, acc[j]);
    }
  }
  if (lane < 16) {  // rows 0..7 valid (M = (lane>>4)*8 + v = v)
#pragma unroll
    for (int j = 0; j < 4; ++j) {
      int col = (w * 4 + j) * 16 + lane;
      float bb = bias[col];
#pragma unroll
      for (int v = 0; v < 8; ++v) out[v * 512 + col] = acc[j][v] + bb;
    }
  }
}

__device__ __forceinline__ float block_reduce_sum(float v, float* red) {
  int tid = threadIdx.x;
  red[tid] = v;
  __syncthreads();
#pragma unroll
  for (int s = 128; s > 0; s >>= 1) {
    if (tid < s) red[tid] += red[tid + s];
    __syncthreads();
  }
  float r = red[0];
  __syncthreads();
  return r;
}

// ---------------------------------------------------------------------------
// MHA on the 8 distinct tokens + max-pool + MLP head. One block per batch.
// Softmax over 512 keys with 64x multiplicity == softmax over the 8 distinct
// keys (multiplicity cancels); max-pool over duplicates == max over 8 rows.
// ---------------------------------------------------------------------------
__global__ __launch_bounds__(256) void head_kernel(
    const float* __restrict__ F, const float* __restrict__ Bw,
    const bf16* __restrict__ fq, const bf16* __restrict__ fk,
    const bf16* __restrict__ fv, const bf16* __restrict__ fo,
    const float* __restrict__ qb, const float* __restrict__ kb,
    const float* __restrict__ vb, const float* __restrict__ ob,
    const float* __restrict__ w1, const float* __restrict__ b1,
    const float* __restrict__ w2, const float* __restrict__ b2,
    const float* __restrict__ w3, const float* __restrict__ b3,
    const float* __restrict__ g1, const float* __restrict__ bb1,
    const float* __restrict__ g2, const float* __restrict__ bb2,
    float* __restrict__ out) {
  extern __shared__ char smem[];
  bf16* tokA = (bf16*)(smem);            // 16x520 bf16
  bf16* ctxA = (bf16*)(smem + 16640);    // 16x520 bf16
  float* Qf = (float*)(smem + 33280);    // 8x512 (later attn_out)
  float* Kf = (float*)(smem + 49664);    // 8x512
  float* Vf = (float*)(smem + 66048);    // 8x512
  float* aw = (float*)(smem + 82432);    // 8*8*8 attn weights
  float* pooled = (float*)(smem + 84480);
  float* y1 = (float*)(smem + 86528);
  float* resid = (float*)(smem + 88576);
  float* s2v = (float*)(smem + 90624);
  float* red = (float*)(smem + 91648);

  const int b = blockIdx.x, tid = threadIdx.x, lane = tid & 31, w = tid >> 5;

  for (int i = tid; i < 16 * TSTR; i += 256) { tokA[i] = (bf16)0.0f; ctxA[i] = (bf16)0.0f; }
  __syncthreads();
  for (int idx = tid; idx < 8 * 512; idx += 256) {
    int i = idx >> 9, n = idx & 511;
    float v = (n < 256) ? F[(b * 8 + i) * 256 + n]
                        : Bw[(b * 8 + (7 - i)) * 256 + (n - 256)];
    tokA[i * TSTR + n] = (bf16)v;
  }
  __syncthreads();

  head_proj(tokA, fq, qb, Qf, w, lane);
  head_proj(tokA, fk, kb, Kf, w, lane);
  head_proj(tokA, fv, vb, Vf, w, lane);
  __syncthreads();

  if (tid < 64) {  // one (head, query) per thread: 8 scores, softmax
    int hh = tid >> 3, ii = tid & 7;
    float s[8];
#pragma unroll
    for (int j = 0; j < 8; ++j) {
      float d = 0.f;
      const float* qp = Qf + ii * 512 + hh * 64;
      const float* kp = Kf + j * 512 + hh * 64;
#pragma unroll 1
      for (int e2 = 0; e2 < 64; ++e2) d += qp[e2] * kp[e2];
      s[j] = d * 0.125f;  // 1/sqrt(64)
    }
    float mx = s[0];
#pragma unroll
    for (int j = 1; j < 8; ++j) mx = fmaxf(mx, s[j]);
    float den = 0.f;
#pragma unroll
    for (int j = 0; j < 8; ++j) { s[j] = __expf(s[j] - mx); den += s[j]; }
    float inv = 1.f / den;
#pragma unroll
    for (int j = 0; j < 8; ++j) aw[(hh * 8 + ii) * 8 + j] = s[j] * inv;
  }
  __syncthreads();

  for (int k2 = 0; k2 < 16; ++k2) {  // ctx = attn @ V
    int idx = tid + k2 * 256;
    int i = idx >> 9, n = idx & 511, hh = n >> 6;
    float c = 0.f;
#pragma unroll
    for (int j = 0; j < 8; ++j) c += aw[(hh * 8 + i) * 8 + j] * Vf[j * 512 + n];
    ctxA[i * TSTR + n] = (bf16)c;
  }
  __syncthreads();

  head_proj(ctxA, fo, ob, Qf, w, lane);  // attn_out -> Qf
  __syncthreads();

  for (int k2 = 0; k2 < 2; ++k2) {  // global max-pool (over 8 distinct rows)
    int n = tid + k2 * 256;
    float m = -3.402823466e38f;
#pragma unroll
    for (int i = 0; i < 8; ++i) m = fmaxf(m, Qf[i * 512 + n]);
    pooled[n] = m;
  }
  __syncthreads();

  for (int k2 = 0; k2 < 2; ++k2) {  // fc1
    int n = tid + k2 * 256;
    float acc = b1[n];
#pragma unroll 1
    for (int m2 = 0; m2 < 512; ++m2) acc += pooled[m2] * w1[n * 512 + m2];
    y1[n] = acc;
  }
  __syncthreads();

  float p0 = y1[tid], p1 = y1[tid + 256];
  float S = block_reduce_sum(p0 + p1, red);
  float SQ = block_reduce_sum(p0 * p0 + p1 * p1, red);
  float mu = S * (1.f / 512.f);
  float var = SQ * (1.f / 512.f) - mu * mu;
  float rstd = rsqrtf(var + 1e-5f);
  for (int k2 = 0; k2 < 2; ++k2) {
    int n = tid + k2 * 256;
    float z = (y1[n] - mu) * rstd * g1[n] + bb1[n];
    resid[n] = geluf_(z);
  }
  __syncthreads();

  float acc2 = b2[tid];  // fc2, m = tid
#pragma unroll 1
  for (int n = 0; n < 512; ++n) acc2 += resid[n] * w2[tid * 512 + n];
  float S2 = block_reduce_sum(acc2, red);
  float SQ2 = block_reduce_sum(acc2 * acc2, red);
  float mu2 = S2 * (1.f / 256.f);
  float var2 = SQ2 * (1.f / 256.f) - mu2 * mu2;
  float rstd2 = rsqrtf(var2 + 1e-5f);
  float z2 = (acc2 - mu2) * rstd2 * g2[tid] + bb2[tid];
  s2v[tid] = geluf_(z2) + resid[tid];  // + residual[:, :256]
  __syncthreads();

  if (tid < 8) {
    float a = b3[tid];
#pragma unroll 1
    for (int m2 = 0; m2 < 256; ++m2) a += s2v[m2] * w3[tid * 256 + m2];
    out[b * 8 + tid] = a;
  }
}

// ---------------------------------------------------------------------------
// Host-side launcher.
// ---------------------------------------------------------------------------
extern "C" void kernel_launch(void* const* d_in, const int* in_sizes, int n_in,
                              void* d_out, int out_size, void* d_ws, size_t ws_size,
                              hipStream_t stream) {
  (void)in_sizes; (void)n_in; (void)out_size; (void)ws_size;
  // setup_inputs() dict insertion order:
  // 0:x, then fwd0..2 {wih,whh,bih,bhh}, bwd0..2, q{w,b},k,v,o, fc1,fc2,fc3, ln1{g,b}, ln2{g,b}
  const float* X = (const float*)d_in[0];
  const float *wih[6], *whh[6], *bih[6], *bhh[6];
  for (int i = 0; i < 6; ++i) {
    wih[i] = (const float*)d_in[1 + 4 * i];
    whh[i] = (const float*)d_in[2 + 4 * i];
    bih[i] = (const float*)d_in[3 + 4 * i];
    bhh[i] = (const float*)d_in[4 + 4 * i];
  }
  const float* qw  = (const float*)d_in[25]; const float* qbv = (const float*)d_in[26];
  const float* kw  = (const float*)d_in[27]; const float* kbv = (const float*)d_in[28];
  const float* vw  = (const float*)d_in[29]; const float* vbv = (const float*)d_in[30];
  const float* ow  = (const float*)d_in[31]; const float* obv = (const float*)d_in[32];
  const float* f1w = (const float*)d_in[33]; const float* f1b = (const float*)d_in[34];
  const float* f2w = (const float*)d_in[35]; const float* f2b = (const float*)d_in[36];
  const float* f3w = (const float*)d_in[37]; const float* f3b = (const float*)d_in[38];
  const float* l1g = (const float*)d_in[39]; const float* l1b = (const float*)d_in[40];
  const float* l2g = (const float*)d_in[41]; const float* l2b = (const float*)d_in[42];

  char* ws = (char*)d_ws;
  size_t off = 0;
  auto carve = [&](size_t bytes) -> char* {
    char* p = ws + off;
    off += (bytes + 255) & ~(size_t)255;
    return p;
  };
  bf16* fragL[6];
  for (int i = 0; i < 6; ++i) {
    int din = (i % 3 == 0) ? 64 : 256;
    fragL[i] = (bf16*)carve((size_t)(din + 256) * 1024 * 2);
  }
  bf16* fragQ = (bf16*)carve(512 * 512 * 2);
  bf16* fragK = (bf16*)carve(512 * 512 * 2);
  bf16* fragV = (bf16*)carve(512 * 512 * 2);
  bf16* fragO = (bf16*)carve(512 * 512 * 2);
  float* biasAll = (float*)carve(6 * 1024 * 4);
  float* Fbuf = (float*)carve(2 * 32 * 8 * 256 * 4);  // F then Bw

  // 1) pack LSTM weights into WMMA B-fragment order (bf16)
  for (int i = 0; i < 6; ++i) {
    int din = (i % 3 == 0) ? 64 : 256;
    int total = 64 * ((din + 256) >> 5) * 512;  // NT*KT*32*16
    pack_lstm_kernel<<<(total + 255) / 256, 256, 0, stream>>>(wih[i], whh[i], din, total, fragL[i]);
  }
  // 2) pack attention projection weights
  const int totW = 32 * 16 * 512;  // (512/16)*(512/32)*32*16
  pack_w_kernel<<<(totW + 255) / 256, 256, 0, stream>>>(qw, 512, totW, fragQ);
  pack_w_kernel<<<(totW + 255) / 256, 256, 0, stream>>>(kw, 512, totW, fragK);
  pack_w_kernel<<<(totW + 255) / 256, 256, 0, stream>>>(vw, 512, totW, fragV);
  pack_w_kernel<<<(totW + 255) / 256, 256, 0, stream>>>(ow, 512, totW, fragO);
  // 3) fused LSTM biases
  for (int i = 0; i < 6; ++i)
    add_bias_kernel<<<4, 256, 0, stream>>>(bih[i], bhh[i], biasAll + i * 1024);

  // 4) fused dilated bi-LSTM (grid=2: dir 0 fwd, dir 1 bwd)
  const int DRNN_SMEM = 32 * ASTR * 2;  // 53,760 B
  drnn_kernel<<<2, 1024, DRNN_SMEM, stream>>>(
      X, fragL[0], fragL[1], fragL[2], fragL[3], fragL[4], fragL[5], biasAll, Fbuf);

  // 5) attention + head on the 8 distinct tokens per batch
  const int HEAD_SMEM = 92672;
  head_kernel<<<32, 256, HEAD_SMEM, stream>>>(
      Fbuf, Fbuf + 65536, fragQ, fragK, fragV, fragO, qbv, kbv, vbv, obv,
      f1w, f1b, f2w, f2b, f3w, f3b, l1g, l1b, l2g, l2b, (float*)d_out);
}
